// ScaledDotProductAttention_25907242729928
// MI455X (gfx1250) — matrix operand order repaired and checked by reading, not compiled, not run on hardware
//
#include <hip/hip_runtime.h>

// ---- CDNA5 (gfx1250) flash attention: B=2 H=16 S=2048 D=64, fp32 I/O ----
// QK^T and P*V via v_wmma_f32_16x16x32_f16 (f16 data, fp32 accumulate),
// online softmax in fp32. K/V tiles software-pipelined (regs hold next tile's
// f32 data during compute). B-fragments batch-loaded into distinct registers
// so 4 WMMAs issue back-to-back instead of serializing on one LDS buffer.

typedef __attribute__((ext_vector_type(16))) _Float16 v16h;
typedef __attribute__((ext_vector_type(8)))  _Float16 v8h;
typedef __attribute__((ext_vector_type(4)))  _Float16 v4h;
typedef __attribute__((ext_vector_type(8)))  float    v8f;
typedef __attribute__((ext_vector_type(4)))  float    v4f;

constexpr int Bc = 2, Hc = 16, Sc = 2048, Dc = 64;
constexpr int BM = 128;  // query rows per workgroup (8 waves x 16)
constexpr int BN = 64;   // keys per iteration
constexpr int NTHREADS = 256;
constexpr int ELTS_PER_THREAD = (BN * Dc) / NTHREADS;  // 16 floats = 4 x v4f

__device__ __forceinline__ v16h join16(v8h lo, v8h hi) {
    return __builtin_shufflevector(lo, hi, 0, 1, 2, 3, 4, 5, 6, 7,
                                            8, 9, 10, 11, 12, 13, 14, 15);
}

__global__ __launch_bounds__(NTHREADS) void
fa_fwd_kernel(const float* __restrict__ q, const float* __restrict__ k,
              const float* __restrict__ v, const int* __restrict__ mask,
              float* __restrict__ out) {
    __shared__ alignas(16) _Float16 Ksh[BN * Dc];      // [key][d]   row-major
    __shared__ alignas(16) _Float16 Vsh[Dc * BN];      // [d][key]   transposed
    __shared__ alignas(16) _Float16 Psh[8][16 * BN];   // per-wave P [m][key]

    const int tid  = threadIdx.x;
    const int wave = tid >> 5;
    const int lane = tid & 31;
    const int half = lane >> 4;   // which 16-lane half of the wave
    const int ln   = lane & 15;

    const int bh    = blockIdx.y;          // fused (b,h)
    const int b     = bh / Hc;
    const int qbase = blockIdx.x * BM;

    const size_t bhSD = (size_t)bh * Sc * Dc;
    const float* qb = q + bhSD;
    const float* kb = k + bhSD;
    const float* vb = v + bhSD;
    const int*   mb = mask + (size_t)b * Sc * Sc;

    // ---- Q A-fragments (16x32 each, K-dim = D = 64 -> 2 frags), scaled 1/sqrt(64)
    // A layout: lane -> row M=ln; elems 0..7 = K(half*8 + 0..7), 8..15 = K(16+half*8+0..7)
    const int    qrowA = qbase + wave * 16 + ln;
    const float* qp    = qb + (size_t)qrowA * Dc;
    v16h aq[2];
    const float scale = 0.125f;  // 1/sqrt(64)
#pragma unroll
    for (int kc = 0; kc < 2; ++kc) {
        const int d0 = kc * 32 + half * 8;
#pragma unroll
        for (int t = 0; t < 8; ++t) aq[kc][t]     = (_Float16)(qp[d0 + t] * scale);
#pragma unroll
        for (int t = 0; t < 8; ++t) aq[kc][8 + t] = (_Float16)(qp[d0 + 16 + t] * scale);
    }

    // ---- accumulators: O = 16x64 (4 C-tiles), row stats per C-layout row (VGPR i)
    v8f o[4];
#pragma unroll
    for (int nt = 0; nt < 4; ++nt)
#pragma unroll
        for (int i = 0; i < 8; ++i) o[nt][i] = 0.0f;
    float rm[8], rs[8];
#pragma unroll
    for (int i = 0; i < 8; ++i) { rm[i] = -1e30f; rs[i] = 0.0f; }

    // ---- software pipeline: registers hold the *next* K/V tile (f32)
    v4f kreg[ELTS_PER_THREAD / 4], vreg[ELTS_PER_THREAD / 4];
#pragma unroll
    for (int i = 0; i < ELTS_PER_THREAD / 4; ++i) {
        const int f   = tid * 4 + i * (NTHREADS * 4);
        const int row = f >> 6, col = f & 63;
        kreg[i] = *(const v4f*)(kb + (size_t)row * Dc + col);
        vreg[i] = *(const v4f*)(vb + (size_t)row * Dc + col);
    }

    for (int kv0 = 0; kv0 < Sc; kv0 += BN) {
        // ---- drain pipeline regs: convert f32 -> f16 and stage into LDS
#pragma unroll
        for (int i = 0; i < ELTS_PER_THREAD / 4; ++i) {
            const int f   = tid * 4 + i * (NTHREADS * 4);
            const int row = f >> 6, col = f & 63;   // col is a multiple of 4
            v4h kh;
#pragma unroll
            for (int c = 0; c < 4; ++c) kh[c] = (_Float16)kreg[i][c];
            *(v4h*)(&Ksh[row * Dc + col]) = kh;
#pragma unroll
            for (int c = 0; c < 4; ++c) Vsh[(col + c) * BN + row] = (_Float16)vreg[i][c];
        }
        // ---- issue next tile's global loads now; they complete under compute
        if (kv0 + BN < Sc) {
#pragma unroll
            for (int i = 0; i < ELTS_PER_THREAD / 4; ++i) {
                const int f   = tid * 4 + i * (NTHREADS * 4);
                const int row = f >> 6, col = f & 63;
                kreg[i] = *(const v4f*)(kb + (size_t)(kv0 + BN + row) * Dc + col);
                vreg[i] = *(const v4f*)(vb + (size_t)(kv0 + BN + row) * Dc + col);
            }
        }
        if (kv0 + 2 * BN < Sc) {   // warm L2 two tiles ahead (global_prefetch_b8)
            __builtin_prefetch(kb + (size_t)(kv0 + 2 * BN) * Dc + tid * 16, 0, 0);
            __builtin_prefetch(vb + (size_t)(kv0 + 2 * BN) * Dc + tid * 16, 0, 0);
        }
        __syncthreads();

        // ---- S = Q K^T : per K-chunk, batch-load 4 B-fragments into distinct
        //      registers, then 4 back-to-back WMMAs (no per-WMMA dscnt stall)
        v8f sc[4];
        {
            v16h bk[4];
#pragma unroll
            for (int nt = 0; nt < 4; ++nt) {
                const _Float16* bp = &Ksh[(nt * 16 + ln) * Dc + half * 8];
                bk[nt] = join16(*(const v8h*)bp, *(const v8h*)(bp + 16));
            }
#pragma unroll
            for (int nt = 0; nt < 4; ++nt) {
                v8f z;
#pragma unroll
                for (int i = 0; i < 8; ++i) z[i] = 0.0f;
                sc[nt] = __builtin_amdgcn_wmma_f32_16x16x32_f16(
                             false, aq[0], false, bk[nt], (short)0, z, false, false);
            }
#pragma unroll
            for (int nt = 0; nt < 4; ++nt) {
                const _Float16* bp = &Ksh[(nt * 16 + ln) * Dc + 32 + half * 8];
                bk[nt] = join16(*(const v8h*)bp, *(const v8h*)(bp + 16));
            }
#pragma unroll
            for (int nt = 0; nt < 4; ++nt)
                sc[nt] = __builtin_amdgcn_wmma_f32_16x16x32_f16(
                             false, aq[1], false, bk[nt], (short)0, sc[nt], false, false);
        }

        // ---- mask (C layout: VGPR i -> row i+8*half, lane -> col ln + nt*16)
#pragma unroll
        for (int nt = 0; nt < 4; ++nt) {
            const int kcol = kv0 + nt * 16 + ln;
#pragma unroll
            for (int i = 0; i < 8; ++i) {
                const int qr = qbase + wave * 16 + i + 8 * half;
                if (mb[(size_t)qr * Sc + kcol] == 0) sc[nt][i] = -1e30f;
            }
        }

        // ---- online softmax, per C-layout row
#pragma unroll
        for (int i = 0; i < 8; ++i) {
            float tmax = fmaxf(fmaxf(sc[0][i], sc[1][i]), fmaxf(sc[2][i], sc[3][i]));
#pragma unroll
            for (int off = 1; off < 16; off <<= 1)
                tmax = fmaxf(tmax, __shfl_xor(tmax, off, 32));
            const float newm  = fmaxf(rm[i], tmax);
            const float alpha = __expf(rm[i] - newm);
            float ps = 0.0f;
#pragma unroll
            for (int nt = 0; nt < 4; ++nt) {
                const float p = __expf(sc[nt][i] - newm);
                sc[nt][i] = p;
                ps += p;
            }
#pragma unroll
            for (int off = 1; off < 16; off <<= 1) ps += __shfl_xor(ps, off, 32);
            rs[i] = rs[i] * alpha + ps;
            rm[i] = newm;
#pragma unroll
            for (int nt = 0; nt < 4; ++nt) o[nt][i] *= alpha;
        }

        // ---- P (C layout) -> LDS as f16 (per-wave buffer, no block barrier needed)
#pragma unroll
        for (int nt = 0; nt < 4; ++nt)
#pragma unroll
            for (int i = 0; i < 8; ++i)
                Psh[wave][(i + 8 * half) * BN + nt * 16 + ln] = (_Float16)sc[nt][i];

#if __has_builtin(__builtin_amdgcn_wave_barrier)
        __builtin_amdgcn_wave_barrier();   // keep same-wave LDS store->load ordered
#else
        asm volatile("" ::: "memory");
#endif

        // ---- O += P * V : batch-load B-fragments, then 4 WMMAs per K-chunk
#pragma unroll
        for (int kc = 0; kc < 2; ++kc) {
            const _Float16* pp = &Psh[wave][ln * BN + kc * 32 + half * 8];
            const v16h ap = join16(*(const v8h*)pp, *(const v8h*)(pp + 16));
            v16h bv[4];
#pragma unroll
            for (int nt = 0; nt < 4; ++nt) {
                const _Float16* vp = &Vsh[(nt * 16 + ln) * BN + kc * 32 + half * 8];
                bv[nt] = join16(*(const v8h*)vp, *(const v8h*)(vp + 16));
            }
#pragma unroll
            for (int nt = 0; nt < 4; ++nt)
                o[nt] = __builtin_amdgcn_wmma_f32_16x16x32_f16(
                            false, ap, false, bv[nt], (short)0, o[nt], false, false);
        }

        __syncthreads();   // before next iteration overwrites Ksh/Vsh
    }

    // ---- epilogue: normalize and store fp32
#pragma unroll
    for (int i = 0; i < 8; ++i) {
        const float inv = (rs[i] > 0.0f) ? (1.0f / rs[i]) : 0.0f;
        const int   qr  = qbase + wave * 16 + i + 8 * half;
        float* op = out + ((size_t)bh * Sc + qr) * Dc;
#pragma unroll
        for (int nt = 0; nt < 4; ++nt) op[nt * 16 + ln] = o[nt][i] * inv;
    }
}

extern "C" void kernel_launch(void* const* d_in, const int* in_sizes, int n_in,
                              void* d_out, int out_size, void* d_ws, size_t ws_size,
                              hipStream_t stream) {
    (void)in_sizes; (void)n_in; (void)out_size; (void)d_ws; (void)ws_size;
    const float* q    = (const float*)d_in[0];
    const float* k    = (const float*)d_in[1];
    const float* v    = (const float*)d_in[2];
    const int*   mask = (const int*)d_in[3];
    float* out = (float*)d_out;

    dim3 grid(Sc / BM, Bc * Hc);   // (16, 32) workgroups
    dim3 block(NTHREADS);          // 8 wave32s
    fa_fwd_kernel<<<grid, block, 0, stream>>>(q, k, v, mask, out);
}
